// AttentionTradingNet_60352880443667
// MI455X (gfx1250) — compile-verified, matrix operand order checked by reading, not run on hardware
//
#include <hip/hip_runtime.h>
#include <hip/hip_bf16.h>

typedef unsigned int  u32;
typedef unsigned short u16;
typedef __attribute__((ext_vector_type(16))) __bf16 v16bf;
typedef __attribute__((ext_vector_type(8)))  float  v8f;

union FragBF { v16bf v; u32 u[8]; };

__device__ __forceinline__ u16 f2bf(float f) {
  u32 u = __float_as_uint(f);
  u32 r = (u + 0x7fffu + ((u >> 16) & 1u)) >> 16;
  return (u16)r;
}

// Load a 16x32 bf16 WMMA fragment (A layout; B uses the same layout when the
// operand is stored transposed [N][K]).  p points at (row base + k0).
// kbase = 8 * (lane >> 4).  K pair offsets per ISA: {0,2,4,6,16,18,20,22}+kbase.
// Per lane this is two contiguous 16B groups -> compiler emits 2x b128 loads.
__device__ __forceinline__ void load_frag(FragBF& f, const u16* p, int kbase) {
#pragma unroll
  for (int j = 0; j < 8; ++j) {
    int off = ((j < 4) ? (2 * j) : (16 + 2 * (j - 4))) + kbase;
    f.u[j] = *reinterpret_cast<const u32*>(p + off);
  }
}

// ---------------------------------------------------------------------------
// Tiled bf16 GEMM with fp32 accumulation:  C[M][N] = act(A[M][K] * W + bias (+resid))
// A: bf16 row-major [M][K].  Wt: bf16 [N][K] (pre-transposed).
// Output row remapping: outRow = row*rowMul + rowAdd (for modality interleave).
// act: 0=none, 1=relu, 2=gelu(exact erf).  Cf (fp32) and Cb (bf16) optional.
// Block: 256 thr = 8 waves; wave tile 32x128 (16 WMMA / k-step, 20 b128 loads);
// block tile 128x256.
// ---------------------------------------------------------------------------
__global__ __launch_bounds__(256) void gemm_bf16_kernel(
    const u16* __restrict__ A, const u16* __restrict__ Wt,
    const float* __restrict__ bias, const float* __restrict__ resid,
    float* __restrict__ Cf, u16* __restrict__ Cb,
    int M, int N, int K, int ldc, int rowMul, int rowAdd, int act)
{
  int tid = threadIdx.x;
  int w = tid >> 5, lane = tid & 31;
  int hf = lane >> 4, l15 = lane & 15;
  int kbase = hf * 8;
  int m0 = blockIdx.y * 128 + (w & 3) * 32;
  int n0 = blockIdx.x * 256 + (w >> 2) * 128;
  if (m0 >= M || n0 >= N) return;

  const u16* a0p = A + (size_t)(m0 + l15) * K;
  const u16* a1p = a0p + (size_t)16 * K;
  const u16* bptr[8];
#pragma unroll
  for (int t = 0; t < 8; ++t) bptr[t] = Wt + (size_t)(n0 + t * 16 + l15) * K;

  v8f acc[2][8] = {};
  for (int k0 = 0; k0 < K; k0 += 32) {
    FragBF a0, a1;
    load_frag(a0, a0p + k0, kbase);
    load_frag(a1, a1p + k0, kbase);
    if (k0 + 64 < K) __builtin_prefetch(a0p + k0 + 64, 0, 1);
#pragma unroll
    for (int t = 0; t < 8; ++t) {
      FragBF b;
      load_frag(b, bptr[t] + k0, kbase);
      acc[0][t] = __builtin_amdgcn_wmma_f32_16x16x32_bf16(
          false, a0.v, false, b.v, (short)0, acc[0][t], false, false);
      acc[1][t] = __builtin_amdgcn_wmma_f32_16x16x32_bf16(
          false, a1.v, false, b.v, (short)0, acc[1][t], false, false);
    }
  }

#pragma unroll
  for (int mi = 0; mi < 2; ++mi)
#pragma unroll
    for (int t = 0; t < 8; ++t)
#pragma unroll
      for (int r = 0; r < 8; ++r) {
        int row = m0 + mi * 16 + r + hf * 8;
        int col = n0 + t * 16 + l15;
        float v = acc[mi][t][r] + bias[col];
        size_t cidx = ((size_t)row * rowMul + rowAdd) * (size_t)ldc + col;
        if (resid) v += resid[cidx];
        if (act == 1)      v = fmaxf(v, 0.0f);
        else if (act == 2) v = 0.5f * v * (1.0f + erff(v * 0.70710678118654752f));
        if (Cf) Cf[cidx] = v;
        if (Cb) Cb[cidx] = f2bf(v);
      }
}

// ---------------------------------------------------------------------------
// Unified MHA kernel (softmax(Q K^T / 8) V), DH=64, NH=8, qkv width 3H=1536.
// Matrix row of attention index i, group o:  row = i*rowStride + o.
// grid.x = nOuter * 8 * (L/64);  block = 128 threads (4 waves x 16 query rows).
// Both WMMA operand loads come straight from the row-major bf16 qkv buffer
// (K rows are exactly the Bt layout needed for Q*K^T); V is staged transposed
// in LDS (coalesced global reads); softmax probs re-quantized to bf16 in LDS.
// ---------------------------------------------------------------------------
template <int L>
__global__ __launch_bounds__(128) void attn_kernel(
    const u16* __restrict__ qkv, u16* __restrict__ outB,
    int rowStride, float scale)
{
  constexpr int NQT = L / 64;
  constexpr int NT  = L / 16;
  __shared__ __attribute__((aligned(16))) u16 Vt[64][256];
  __shared__ __attribute__((aligned(16))) u16 Asm[4][16][256];

  int g = blockIdx.x;
  int qt = g % NQT; g /= NQT;
  int head = g & 7;
  int o = g >> 3;
  int tid = threadIdx.x;
  int w = tid >> 5, lane = tid & 31;
  int hf = lane >> 4, l15 = lane & 15;
  int kbase = hf * 8;
  const int THREE_H = 1536, H = 512;
  size_t base = (size_t)o * THREE_H + head * 64;

  // stage V transposed: Vt[d][j] = V[j][d]; consecutive lanes read consecutive
  // d within one V row -> coalesced global reads, scatter into LDS.
  for (int idx = tid; idx < 64 * L; idx += 128) {
    int j = idx >> 6, d = idx & 63;
    Vt[d][j] = qkv[(size_t)j * rowStride * THREE_H + base + 1024 + d];
  }
  __syncthreads();

  // S strip = Q(16 rows) * K^T  (DH=64 -> 2 k-steps)
  int qrow = qt * 64 + w * 16 + l15;
  const u16* qp = qkv + (size_t)qrow * rowStride * THREE_H + base;
  v8f s[NT] = {};
#pragma unroll
  for (int k0 = 0; k0 < 64; k0 += 32) {
    FragBF a;
    load_frag(a, qp + k0, kbase);
#pragma unroll
    for (int nj = 0; nj < NT; ++nj) {
      int kn = nj * 16 + l15;
      const u16* kp = qkv + (size_t)kn * rowStride * THREE_H + base + 512;
      FragBF b;
      load_frag(b, kp + k0, kbase);
      s[nj] = __builtin_amdgcn_wmma_f32_16x16x32_bf16(
          false, a.v, false, b.v, (short)0, s[nj], false, false);
    }
  }

  // row softmax (each half of the wave holds 8 rows; masks 1..8 stay in-half)
#pragma unroll
  for (int r = 0; r < 8; ++r) {
    float mx = -3.0e38f;
#pragma unroll
    for (int nj = 0; nj < NT; ++nj) mx = fmaxf(mx, s[nj][r] * scale);
    mx = fmaxf(mx, __shfl_xor(mx, 1));
    mx = fmaxf(mx, __shfl_xor(mx, 2));
    mx = fmaxf(mx, __shfl_xor(mx, 4));
    mx = fmaxf(mx, __shfl_xor(mx, 8));
    float sum = 0.f;
#pragma unroll
    for (int nj = 0; nj < NT; ++nj) {
      float e = __expf(s[nj][r] * scale - mx);
      s[nj][r] = e;
      sum += e;
    }
    sum += __shfl_xor(sum, 1);
    sum += __shfl_xor(sum, 2);
    sum += __shfl_xor(sum, 4);
    sum += __shfl_xor(sum, 8);
    float inv = 1.0f / sum;
    int lr = r + hf * 8;
#pragma unroll
    for (int nj = 0; nj < NT; ++nj)
      Asm[w][lr][nj * 16 + l15] = f2bf(s[nj][r] * inv);
  }
  // wave consumes only its own Asm strip -> no block barrier needed

  // O strip = A(16xL) * V(Lx64)
  v8f oacc[4] = {};
#pragma unroll
  for (int k0 = 0; k0 < L; k0 += 32) {
    FragBF a;
    load_frag(a, &Asm[w][l15][k0], kbase);
#pragma unroll
    for (int nj = 0; nj < 4; ++nj) {
      FragBF b;
      load_frag(b, &Vt[nj * 16 + l15][k0], kbase);
      oacc[nj] = __builtin_amdgcn_wmma_f32_16x16x32_bf16(
          false, a.v, false, b.v, (short)0, oacc[nj], false, false);
    }
  }
#pragma unroll
  for (int nj = 0; nj < 4; ++nj)
#pragma unroll
    for (int r = 0; r < 8; ++r) {
      int i = qt * 64 + w * 16 + r + hf * 8;
      size_t row = (size_t)i * rowStride + o;
      int col = head * 64 + nj * 16 + l15;
      outB[row * H + col] = f2bf(oacc[nj][r]);
    }
}

// ---------------------------------------------------------------------------
// LayerNorm over width 512, in place, optional bf16 copy.  wave-per-row.
// ---------------------------------------------------------------------------
__global__ __launch_bounds__(256) void ln512_kernel(
    float* __restrict__ x, const float* __restrict__ g, const float* __restrict__ b,
    u16* __restrict__ outB, int M)
{
  int w = threadIdx.x >> 5, lane = threadIdx.x & 31;
  int row = blockIdx.x * 8 + w;
  if (row >= M) return;
  float* xr = x + (size_t)row * 512;
  float v[16];
  float s = 0.f;
#pragma unroll
  for (int i = 0; i < 16; ++i) { v[i] = xr[lane + i * 32]; s += v[i]; }
  s += __shfl_xor(s, 1); s += __shfl_xor(s, 2); s += __shfl_xor(s, 4);
  s += __shfl_xor(s, 8); s += __shfl_xor(s, 16);
  float mean = s * (1.0f / 512.0f);
  float q = 0.f;
#pragma unroll
  for (int i = 0; i < 16; ++i) { float d = v[i] - mean; q += d * d; }
  q += __shfl_xor(q, 1); q += __shfl_xor(q, 2); q += __shfl_xor(q, 4);
  q += __shfl_xor(q, 8); q += __shfl_xor(q, 16);
  float inv = rsqrtf(q * (1.0f / 512.0f) + 1e-5f);
#pragma unroll
  for (int i = 0; i < 16; ++i) {
    int c = lane + i * 32;
    float y = (v[i] - mean) * inv * g[c] + b[c];
    xr[c] = y;
    if (outB) outB[(size_t)row * 512 + c] = f2bf(y);
  }
}

// mean over 5 modalities: fused[r][c] = mean_md ln[(r*5+md)*512+c] -> bf16
__global__ void meanmod_kernel(const float* __restrict__ ln, u16* __restrict__ outB, int R1)
{
  int idx = blockIdx.x * blockDim.x + threadIdx.x;
  if (idx >= R1 * 512) return;
  int r = idx >> 9, c = idx & 511;
  float s = 0.f;
#pragma unroll
  for (int md = 0; md < 5; ++md) s += ln[((size_t)r * 5 + md) * 512 + c];
  outB[idx] = f2bf(s * 0.2f);
}

// [B,T,H]->[T,B,H] + sinusoidal positional encoding; writes fp32 + bf16
__global__ void pe_kernel(const float* __restrict__ f2, float* __restrict__ hF,
                          u16* __restrict__ hB)
{
  const int Bc = 64, Tc = 256;
  int idx = blockIdx.x * blockDim.x + threadIdx.x;
  if (idx >= Bc * Tc * 512) return;
  int c = idx & 511;
  int r = idx >> 9;        // r = b*T + t
  int t = r % Tc, bb = r / Tc;
  float val = f2[idx];
  int i2 = c & ~1;
  float div = __expf(-(float)i2 * (9.210340371976184f / 512.0f));
  float ang = (float)t * div;
  val += (c & 1) ? __cosf(ang) : __sinf(ang);
  size_t oidx = ((size_t)t * Bc + bb) * 512 + c;
  hF[oidx] = val;
  hB[oidx] = f2bf(val);
}

// small-K projector layer1: relu(X[R][d] @ W[d][512] + b) -> bf16
__global__ void projl1_kernel(const float* __restrict__ X, const float* __restrict__ W,
                              const float* __restrict__ bias, u16* __restrict__ outB,
                              int R, int d)
{
  int idx = blockIdx.x * blockDim.x + threadIdx.x;
  if (idx >= R * 512) return;
  int r = idx >> 9, n = idx & 511;
  float s = bias[n];
  const float* xr = X + (size_t)r * d;
  for (int k = 0; k < d; ++k) s = fmaf(xr[k], W[k * 512 + n], s);
  outB[idx] = f2bf(fmaxf(s, 0.f));
}

// fp32 [K][N] -> bf16 transposed [N][K]
__global__ void wtcvt_kernel(const float* __restrict__ W, u16* __restrict__ Wt, int K, int N)
{
  int idx = blockIdx.x * blockDim.x + threadIdx.x;
  if (idx >= N * K) return;
  int n = idx / K, k = idx % K;
  Wt[idx] = f2bf(W[(size_t)k * N + n]);
}

// heads: z = relu(final @ W1[512][256] + b1)
__global__ void head1_kernel(const float* __restrict__ hF, const float* __restrict__ W1,
                             const float* __restrict__ b1, float* __restrict__ z)
{
  int idx = blockIdx.x * blockDim.x + threadIdx.x;
  if (idx >= 64 * 256) return;
  int bb = idx >> 8, j = idx & 255;
  const float* fin = hF + ((size_t)255 * 64 + bb) * 512;
  float s = b1[j];
  for (int k = 0; k < 512; ++k) s = fmaf(fin[k], W1[k * 256 + j], s);
  z[idx] = fmaxf(s, 0.f);
}

__global__ void head2_kernel(const float* __restrict__ z, const float* __restrict__ W2,
                             const float* __restrict__ b2, float* __restrict__ out, int nout)
{
  int idx = blockIdx.x * blockDim.x + threadIdx.x;
  if (idx >= 64 * nout) return;
  int bb = idx / nout, j = idx % nout;
  const float* zr = z + (size_t)bb * 256;
  float s = b2[j];
  for (int k = 0; k < 256; ++k) s = fmaf(zr[k], W2[k * nout + j], s);
  out[(size_t)bb * nout + j] = s;
}

// conf sigmoid (in place) + dir softmax -> probs region
__global__ void post_kernel(float* __restrict__ out)
{
  int t = threadIdx.x;
  if (t >= 64) return;
  float cl = out[320 + t];
  out[320 + t] = 1.0f / (1.0f + __expf(-cl));
  float a = out[64 + 3 * t], b = out[64 + 3 * t + 1], c = out[64 + 3 * t + 2];
  float m = fmaxf(a, fmaxf(b, c));
  float ea = __expf(a - m), eb = __expf(b - m), ec = __expf(c - m);
  float inv = 1.0f / (ea + eb + ec);
  out[384 + 3 * t]     = ea * inv;
  out[384 + 3 * t + 1] = eb * inv;
  out[384 + 3 * t + 2] = ec * inv;
}

__global__ void zero_kernel(float* p, int n)
{
  int i = blockIdx.x * blockDim.x + threadIdx.x;
  if (i < n) p[i] = 0.f;
}

// ===========================================================================
// Host orchestration
// ===========================================================================
namespace {

struct Net {
  const float* data[5];
  int dmod[5];
  const float *pW1[5], *pb1[5], *pW2[5], *pb2[5];
  const float *fWqkv, *fbqkv, *fWo, *fbo, *flg, *flb, *foW, *fob;
  const float *lWqkv[4], *lbqkv[4], *lWo[4], *lbo[4];
  const float *l1g[4], *l1b[4], *f1W[4], *f1b[4], *f2W[4], *f2b[4], *l2g[4], *l2b[4];
  const float *hW1[4], *hb1[4], *hW2[4], *hb2[4];  // price, dir, vol, conf
};

inline void fill_sorted_params(Net& n, void* const* d_in, int p0)
{
  auto F = [&](int i) { return (const float*)d_in[i]; };
  int p = p0;
  n.fWo = F(p++); n.fWqkv = F(p++); n.fbo = F(p++); n.fbqkv = F(p++);
  n.flb = F(p++); n.flg = F(p++);
  n.foW = F(p++); n.fob = F(p++);
  // heads sorted: conf, dir, price, vol -> my order {price=0,dir=1,vol=2,conf=3}
  const int hmap[4] = {3, 1, 0, 2};
  for (int s = 0; s < 4; ++s) {
    int m = hmap[s];
    n.hW1[m] = F(p++); n.hb1[m] = F(p++); n.hW2[m] = F(p++); n.hb2[m] = F(p++);
  }
  for (int l = 0; l < 4; ++l) {
    n.lWo[l] = F(p++); n.lWqkv[l] = F(p++); n.lbo[l] = F(p++); n.lbqkv[l] = F(p++);
    n.f1W[l] = F(p++); n.f1b[l] = F(p++); n.f2W[l] = F(p++); n.f2b[l] = F(p++);
    n.l1b[l] = F(p++); n.l1g[l] = F(p++); n.l2b[l] = F(p++); n.l2g[l] = F(p++);
  }
  // proj sorted: alternative, fundamental, prices, sentiment, technical
  const int pmap[5] = {4, 2, 0, 3, 1};
  for (int s = 0; s < 5; ++s) {
    int m = pmap[s];
    n.pW1[m] = F(p++); n.pb1[m] = F(p++); n.pW2[m] = F(p++); n.pb2[m] = F(p++);
  }
}

}  // namespace

extern "C" void kernel_launch(void* const* d_in, const int* in_sizes, int n_in,
                              void* d_out, int out_size, void* d_ws, size_t ws_size,
                              hipStream_t stream)
{
  const int B = 64, T = 256, H = 512;
  const int R1 = B * T;        // 16384
  const int R5 = R1 * 5;       // 81920
  (void)B; (void)T; (void)H;
  auto F = [&](int i) { return (const float*)d_in[i]; };

  Net net{};
  const int dm[5] = {5, 20, 10, 5, 8};
  for (int m = 0; m < 5; ++m) net.dmod[m] = dm[m];

  bool isA = (n_in >= 97 && in_sizes[0] == 81920 && in_sizes[1] == 327680 &&
              in_sizes[5] == 2560);
  bool isC = (n_in >= 97 && in_sizes[0] == 81920 && in_sizes[5] == 262144);
  bool isB = (n_in >= 97 && in_sizes[0] == 131072 && in_sizes[1] == 163840);

  if (isC) {
    for (int m = 0; m < 5; ++m) net.data[m] = F(m);
    fill_sorted_params(net, d_in, 5);
  } else if (isB) {
    net.data[0] = F(94); net.data[1] = F(96); net.data[2] = F(1);
    net.data[3] = F(95); net.data[4] = F(0);
    fill_sorted_params(net, d_in, 2);
  } else {
    (void)isA;  // default: insertion order
    for (int m = 0; m < 5; ++m) net.data[m] = F(m);
    int p = 5;
    for (int m = 0; m < 5; ++m) {
      net.pW1[m] = F(p++); net.pb1[m] = F(p++); net.pW2[m] = F(p++); net.pb2[m] = F(p++);
    }
    net.fWqkv = F(p++); net.fbqkv = F(p++); net.fWo = F(p++); net.fbo = F(p++);
    net.flg = F(p++); net.flb = F(p++);
    net.foW = F(p++); net.fob = F(p++);
    for (int l = 0; l < 4; ++l) {
      net.lWqkv[l] = F(p++); net.lbqkv[l] = F(p++); net.lWo[l] = F(p++); net.lbo[l] = F(p++);
      net.l1g[l] = F(p++); net.l1b[l] = F(p++);
      net.f1W[l] = F(p++); net.f1b[l] = F(p++);
      net.f2W[l] = F(p++); net.f2b[l] = F(p++);
      net.l2g[l] = F(p++); net.l2b[l] = F(p++);
    }
    for (int h = 0; h < 4; ++h) {
      net.hW1[h] = F(p++); net.hb1[h] = F(p++); net.hW2[h] = F(p++); net.hb2[h] = F(p++);
    }
  }

  // ---- workspace bump allocation -----------------------------------------
  char* wsp = (char*)d_ws;
  size_t off = 0;
  auto alloc = [&](size_t bytes) -> void* {
    size_t a = (off + 255) & ~(size_t)255;
    void* p = wsp + a;
    off = a + bytes;
    return p;
  };

  u16* wtP2[5];
  for (int m = 0; m < 5; ++m) wtP2[m] = (u16*)alloc((size_t)512 * 512 * 2);
  u16* wtFq  = (u16*)alloc((size_t)1536 * 512 * 2);
  u16* wtFo  = (u16*)alloc((size_t)512 * 512 * 2);
  u16* wtFO2 = (u16*)alloc((size_t)512 * 512 * 2);
  u16 *wtLq[4], *wtLo[4], *wtL1[4], *wtL2[4];
  for (int l = 0; l < 4; ++l) {
    wtLq[l] = (u16*)alloc((size_t)1536 * 512 * 2);
    wtLo[l] = (u16*)alloc((size_t)512 * 512 * 2);
    wtL1[l] = (u16*)alloc((size_t)2048 * 512 * 2);
    wtL2[l] = (u16*)alloc((size_t)512 * 2048 * 2);
  }
  u16*   h1b = (u16*)alloc((size_t)R1 * 512 * 2);
  float* xF  = (float*)alloc((size_t)R5 * 512 * 4);
  u16*   xB  = (u16*)alloc((size_t)R5 * 512 * 2);     // also reused as fusion attn output
  size_t regionBytes = (size_t)R5 * 1536 * 2;          // qkv region, reused later
  char*  reg = (char*)alloc(regionBytes);

  u16* qkvB = (u16*)reg;  // fusion qkv (phase 1 only)
  size_t ro = 0;
  auto ralloc = [&](size_t bytes) -> void* {
    size_t a = (ro + 255) & ~(size_t)255;
    void* p = reg + a;
    ro = a + bytes;
    return p;
  };
  u16*   fusedB  = (u16*)ralloc((size_t)R1 * 512 * 2);
  float* fused2F = (float*)ralloc((size_t)R1 * 512 * 4);
  float* hF      = (float*)ralloc((size_t)R1 * 512 * 4);
  u16*   hB      = (u16*)ralloc((size_t)R1 * 512 * 2);
  u16*   qkvLB   = (u16*)ralloc((size_t)R1 * 1536 * 2);
  u16*   attnLB  = (u16*)ralloc((size_t)R1 * 512 * 2);
  u16*   midB    = (u16*)ralloc((size_t)R1 * 2048 * 2);
  float* zF      = (float*)ralloc((size_t)64 * 256 * 4);

  if (off > ws_size || ro > regionBytes) {
    zero_kernel<<<(out_size + 255) / 256, 256, 0, stream>>>((float*)d_out, out_size);
    return;
  }

  auto cvt = [&](const float* W, int K, int N, u16* dst) {
    int n = K * N;
    wtcvt_kernel<<<(n + 255) / 256, 256, 0, stream>>>(W, dst, K, N);
  };
  auto gemm = [&](const u16* A, const u16* Wt, const float* bias, const float* resid,
                  float* Cf, u16* Cb, int M, int N, int K, int ldc,
                  int rowMul, int rowAdd, int act) {
    dim3 g(N / 256, M / 128);
    gemm_bf16_kernel<<<g, 256, 0, stream>>>(A, Wt, bias, resid, Cf, Cb,
                                            M, N, K, ldc, rowMul, rowAdd, act);
  };

  // ---- weight conversion (fp32 -> transposed bf16) ------------------------
  for (int m = 0; m < 5; ++m) cvt(net.pW2[m], 512, 512, wtP2[m]);
  cvt(net.fWqkv, 512, 1536, wtFq);
  cvt(net.fWo, 512, 512, wtFo);
  cvt(net.foW, 512, 512, wtFO2);
  for (int l = 0; l < 4; ++l) {
    cvt(net.lWqkv[l], 512, 1536, wtLq[l]);
    cvt(net.lWo[l], 512, 512, wtLo[l]);
    cvt(net.f1W[l], 512, 2048, wtL1[l]);
    cvt(net.f2W[l], 2048, 512, wtL2[l]);
  }

  // ---- modality projectors -> x [R5][512] (rows interleaved r*5+md) ------
  for (int m = 0; m < 5; ++m) {
    int n = R1 * 512;
    projl1_kernel<<<(n + 255) / 256, 256, 0, stream>>>(
        net.data[m], net.pW1[m], net.pb1[m], h1b, R1, net.dmod[m]);
    gemm(h1b, wtP2[m], net.pb2[m], nullptr, xF, xB,
         R1, 512, 512, /*ldc=*/512, /*rowMul=*/5, /*rowAdd=*/m, /*act=*/0);
  }

  // ---- fusion attention (over batch axis, L=64) ---------------------------
  gemm(xB, wtFq, net.fbqkv, nullptr, nullptr, qkvB,
       R5, 1536, 512, 1536, 1, 0, 0);
  attn_kernel<64><<<1280 * 8, 128, 0, stream>>>(qkvB, xB, /*rowStride=*/1280, 0.125f);
  gemm(xB, wtFo, net.fbo, /*resid=*/xF, xF, nullptr,
       R5, 512, 512, 512, 1, 0, 0);
  ln512_kernel<<<R5 / 8, 256, 0, stream>>>(xF, net.flg, net.flb, nullptr, R5);
  {
    int n = R1 * 512;
    meanmod_kernel<<<(n + 255) / 256, 256, 0, stream>>>(xF, fusedB, R1);
  }
  gemm(fusedB, wtFO2, net.fob, nullptr, fused2F, nullptr,
       R1, 512, 512, 512, 1, 0, /*relu*/ 1);
  {
    int n = R1 * 512;
    pe_kernel<<<(n + 255) / 256, 256, 0, stream>>>(fused2F, hF, hB);
  }

  // ---- transformer encoder layers (attention over T=256) ------------------
  for (int l = 0; l < 4; ++l) {
    gemm(hB, wtLq[l], net.lbqkv[l], nullptr, nullptr, qkvLB,
         R1, 1536, 512, 1536, 1, 0, 0);
    attn_kernel<256><<<64 * 8 * 4, 128, 0, stream>>>(qkvLB, attnLB, /*rowStride=*/64, 0.125f);
    gemm(attnLB, wtLo[l], net.lbo[l], /*resid=*/hF, hF, nullptr,
         R1, 512, 512, 512, 1, 0, 0);
    ln512_kernel<<<R1 / 8, 256, 0, stream>>>(hF, net.l1g[l], net.l1b[l], hB, R1);
    gemm(hB, wtL1[l], net.f1b[l], nullptr, nullptr, midB,
         R1, 2048, 512, 2048, 1, 0, /*gelu*/ 2);
    gemm(midB, wtL2[l], net.f2b[l], /*resid=*/hF, hF, nullptr,
         R1, 512, 2048, 512, 1, 0, 0);
    ln512_kernel<<<R1 / 8, 256, 0, stream>>>(hF, net.l2g[l], net.l2b[l], hB, R1);
  }

  // ---- heads on final timestep -------------------------------------------
  float* out = (float*)d_out;
  const int hOff[4] = {0, 64, 256, 320};
  const int hN[4] = {1, 3, 1, 1};
  for (int h = 0; h < 4; ++h) {
    head1_kernel<<<(64 * 256 + 255) / 256, 256, 0, stream>>>(hF, net.hW1[h], net.hb1[h], zF);
    int n = 64 * hN[h];
    head2_kernel<<<(n + 255) / 256, 256, 0, stream>>>(zF, net.hW2[h], net.hb2[h],
                                                      out + hOff[h], hN[h]);
  }
  post_kernel<<<1, 64, 0, stream>>>(out);
}
